// IntentionHeads_78288663872370
// MI455X (gfx1250) — matrix operand order repaired and checked by reading, not compiled, not run on hardware
//
#include <hip/hip_runtime.h>
#include <hip/hip_bf16.h>
#include <math.h>

typedef __attribute__((ext_vector_type(16))) __bf16 v16bf;
typedef __attribute__((ext_vector_type(8)))  __bf16 v8bf;
typedef __attribute__((ext_vector_type(8)))  float  v8f;
typedef __attribute__((ext_vector_type(4)))  float  v4f;

#define D_IN   512
#define H_HID  256
#define ROWS   2                          // 16-token row-tiles per WG
#define TOK_WG (16 * ROWS)                // 32 tokens per WG
#define W1_ELEMS (H_HID * D_IN)           // per head: 131072
#define W1_TOTAL (2 * W1_ELEMS)           // both heads: 262144

// ---------------------------------------------------------------------------
// Prep: split fp32 W1 (both heads) into bf16 hi + lo residual, row-major
// [head][neuron][k] so B-matrix fragments are contiguous per lane.
// ---------------------------------------------------------------------------
__global__ void ih_prep_w1(const float* __restrict__ vW1,
                           const float* __restrict__ pW1,
                           __bf16* __restrict__ whi,
                           __bf16* __restrict__ wlo) {
    int idx = blockIdx.x * 256 + threadIdx.x;
    if (idx >= W1_TOTAL) return;
    float w = (idx < W1_ELEMS) ? vW1[idx] : pW1[idx - W1_ELEMS];
    __bf16 hi = (__bf16)w;
    __bf16 lo = (__bf16)(w - (float)hi);
    whi[idx] = hi;
    wlo[idx] = lo;
}

__device__ __forceinline__ float gelu_exact(float v) {
    return 0.5f * v * (1.0f + erff(v * 0.70710678118654752f));
}

// ---------------------------------------------------------------------------
// Main: 8 waves / WG, 32 tokens / WG (2 row-tiles of 16 per wave).
//   Phase 1: stage x tile in LDS as bf16 hi/lo (non-temporal global loads)
//   Phase 2: layer-1 GEMM, bf16 WMMA 3-product split; each B fragment pair
//            feeds 6 WMMAs (2 row-tiles) to amortize L2 weight traffic
//   Phase 3: GELU -> hidden tile in LDS (aliases x region)
//   Phase 4: tiny layer-2 dots + type masking + mask outputs
// ---------------------------------------------------------------------------
__global__ __launch_bounds__(256, 1) void ih_heads_kernel(
    const float* __restrict__ x,        // [B*N, 512]
    const int*   __restrict__ types,    // [B*N]
    const __bf16* __restrict__ whi,     // [2][256][512]
    const __bf16* __restrict__ wlo,     // [2][256][512]
    const float* __restrict__ vb1,      // [256]
    const float* __restrict__ pb1,      // [256]
    const float* __restrict__ vW2,      // [6][256]
    const float* __restrict__ vb2,      // [6]
    const float* __restrict__ pW2,      // [2][256]
    const float* __restrict__ pb2,      // [2]
    float* __restrict__ out_logits,     // [B*N, 6]
    float* __restrict__ out_vmask,      // [B*N]
    float* __restrict__ out_pmask)      // [B*N]
{
    // 64 KB LDS: first bf16 x_hi[32][512] + x_lo[32][512],
    // then reused (after barrier) as fp32 h[32][512].
    __shared__ __align__(16) unsigned char smem[TOK_WG * D_IN * 4];
    __bf16* xhi  = (__bf16*)smem;                  // 32*512 bf16 (32 KB)
    __bf16* xlo  = xhi + TOK_WG * D_IN;            // 32*512 bf16 (32 KB)
    float*  hmem = (float*)smem;                   // 32*512 f32 (aliases)

    const int tid  = threadIdx.x;
    const int tok0 = blockIdx.x * TOK_WG;

    // ---- Phase 1: cooperative load + bf16 split of the 32x512 x tile ------
    // x is streamed exactly once -> non-temporal to keep weights L2-resident.
    const float* xg = x + (size_t)tok0 * D_IN;
    #pragma unroll
    for (int i = 0; i < (TOK_WG * D_IN) / 256; ++i) {
        int f = tid + i * 256;                     // coalesced
        float w = __builtin_nontemporal_load(&xg[f]);
        __bf16 hi = (__bf16)w;
        xhi[f] = hi;
        xlo[f] = (__bf16)(w - (float)hi);
    }
    __syncthreads();

    const int wave = tid >> 5;
    const int lane = tid & 31;
    const int r    = lane & 15;    // A row / B col / C col within 16-tile
    const int half = lane >> 4;    // K half selector

    // Accumulators: [row-tile][N-tile]; waves 0-3 vehicle, 4-7 pedestrian.
    v8f acc[ROWS][4] = {};

    // per-tile weight base: head*131072 + neuron*512, neuron = (tglob&15)*16 + r
    size_t wbase[4];
    #pragma unroll
    for (int tt = 0; tt < 4; ++tt) {
        int tglob = wave * 4 + tt;                 // 0..31
        wbase[tt] = (size_t)(tglob >> 4) * W1_ELEMS
                  + (size_t)(((tglob & 15) * 16) + r) * D_IN;
    }

    #pragma unroll
    for (int c = 0; c < D_IN / 32; ++c) {          // 16 K-chunks of 32
        const int k0 = c * 32;
        // A fragments for both row-tiles from LDS (K order chosen here:
        // elems0-15 <- K=k0+half*16 .. +15, contiguous, matching B)
        v16bf a_hi[ROWS], a_lo[ROWS];
        #pragma unroll
        for (int rt = 0; rt < ROWS; ++rt) {
            const __bf16* xh = xhi + (rt * 16 + r) * D_IN;
            const __bf16* xl = xlo + (rt * 16 + r) * D_IN;
            v8bf h0 = *(const v8bf*)&xh[k0 + half * 16];
            v8bf h1 = *(const v8bf*)&xh[k0 + half * 16 + 8];
            v8bf l0 = *(const v8bf*)&xl[k0 + half * 16];
            v8bf l1 = *(const v8bf*)&xl[k0 + half * 16 + 8];
            #pragma unroll
            for (int j = 0; j < 8; ++j) {
                a_hi[rt][j] = h0[j]; a_hi[rt][8 + j] = h1[j];
                a_lo[rt][j] = l0[j]; a_lo[rt][8 + j] = l1[j];
            }
        }
        const int kb = k0 + half * 16;             // B: 16 contiguous K per lane
        #pragma unroll
        for (int tt = 0; tt < 4; ++tt) {
            v16bf b_hi = *(const v16bf*)(whi + wbase[tt] + kb);
            v16bf b_lo = *(const v16bf*)(wlo + wbase[tt] + kb);
            #pragma unroll
            for (int rt = 0; rt < ROWS; ++rt) {
                v8f a = acc[rt][tt];
                a = __builtin_amdgcn_wmma_f32_16x16x32_bf16(
                        false, a_hi[rt], false, b_hi, (short)0, a, false, false);
                a = __builtin_amdgcn_wmma_f32_16x16x32_bf16(
                        false, a_hi[rt], false, b_lo, (short)0, a, false, false);
                a = __builtin_amdgcn_wmma_f32_16x16x32_bf16(
                        false, a_lo[rt], false, b_hi, (short)0, a, false, false);
                acc[rt][tt] = a;
            }
        }
    }

    // bias per lane (C/D layout: whole lane shares column N = r)
    float bias[4];
    #pragma unroll
    for (int tt = 0; tt < 4; ++tt) {
        int tglob = wave * 4 + tt;
        int n = ((tglob & 15) * 16) + r;
        bias[tt] = (tglob >= 16 ? pb1 : vb1)[n];
    }

    __syncthreads();   // everyone done reading xhi/xlo before aliasing as hmem

    // ---- Phase 3: GELU + store hidden tile (f32) ---------------------------
    #pragma unroll
    for (int rt = 0; rt < ROWS; ++rt) {
        #pragma unroll
        for (int tt = 0; tt < 4; ++tt) {
            int tglob = wave * 4 + tt;
            int hcol = tglob * 16 + r;             // 0..255 veh, 256..511 ped
            #pragma unroll
            for (int v = 0; v < 8; ++v) {
                int m = rt * 16 + v + half * 8;    // token row within WG tile
                hmem[m * D_IN + hcol] = gelu_exact(acc[rt][tt][v] + bias[tt]);
            }
        }
    }
    __syncthreads();

    // ---- Phase 4: layer 2 (tiny) + masking + mask outputs ------------------
    if (tid < TOK_WG * 6) {                        // 192 threads: one (t,o) each
        int t = tid / 6, o = tid % 6;
        int token = tok0 + t;
        int ty = types[token];
        float res = 0.0f;
        if (ty == 1) {
            const v4f* hrow = (const v4f*)(hmem + t * D_IN);        // veh hidden
            const v4f* wrow = (const v4f*)(vW2 + o * H_HID);
            float s = vb2[o];
            #pragma unroll 8
            for (int j = 0; j < H_HID / 4; ++j) {
                v4f h4 = hrow[j], w4 = wrow[j];
                s += h4.x * w4.x + h4.y * w4.y + h4.z * w4.z + h4.w * w4.w;
            }
            res = s;
        } else if (ty == 2 && o < 2) {
            const v4f* hrow = (const v4f*)(hmem + t * D_IN + H_HID); // ped hidden
            const v4f* wrow = (const v4f*)(pW2 + o * H_HID);
            float s = pb2[o];
            #pragma unroll 8
            for (int j = 0; j < H_HID / 4; ++j) {
                v4f h4 = hrow[j], w4 = wrow[j];
                s += h4.x * w4.x + h4.y * w4.y + h4.z * w4.z + h4.w * w4.w;
            }
            res = s;
        }
        __builtin_nontemporal_store(res, &out_logits[(size_t)token * 6 + o]);
    } else if (tid < TOK_WG * 6 + TOK_WG) {        // 32 threads: masks
        int t = tid - TOK_WG * 6;
        int token = tok0 + t;
        int ty = types[token];
        __builtin_nontemporal_store((ty == 1) ? 1.0f : 0.0f, &out_vmask[token]);
        __builtin_nontemporal_store((ty == 2) ? 1.0f : 0.0f, &out_pmask[token]);
    }
}

extern "C" void kernel_launch(void* const* d_in, const int* in_sizes, int n_in,
                              void* d_out, int out_size, void* d_ws, size_t ws_size,
                              hipStream_t stream) {
    const float* x   = (const float*)d_in[0];
    const int*   ty  = (const int*)  d_in[1];
    const float* vW1 = (const float*)d_in[2];
    const float* vb1 = (const float*)d_in[3];
    const float* vW2 = (const float*)d_in[4];
    const float* vb2 = (const float*)d_in[5];
    const float* pW1 = (const float*)d_in[6];
    const float* pb1 = (const float*)d_in[7];
    const float* pW2 = (const float*)d_in[8];
    const float* pb2 = (const float*)d_in[9];

    __bf16* whi = (__bf16*)d_ws;                 // 262144 bf16 = 512 KB
    __bf16* wlo = whi + W1_TOTAL;                // another 512 KB

    const int ntok = in_sizes[1];                // B*N = 65536
    float* out_logits = (float*)d_out;
    float* out_vmask  = out_logits + (size_t)ntok * 6;
    float* out_pmask  = out_vmask + ntok;

    ih_prep_w1<<<(W1_TOTAL + 255) / 256, 256, 0, stream>>>(vW1, pW1, whi, wlo);
    ih_heads_kernel<<<ntok / TOK_WG, 256, 0, stream>>>(
        x, ty, whi, wlo, vb1, pb1, vW2, vb2, pW2, pb2,
        out_logits, out_vmask, out_pmask);
}
